// GRULayer_16587163698033
// MI455X (gfx1250) — matrix operand order repaired and checked by reading, not compiled, not run on hardware
//
#include <hip/hip_runtime.h>

#define B_ 64
#define T_ 1024
#define I_ 512
#define H_ 512
#define LDA 520   // padded LDS row stride in ushorts: 1040B % 256B != 0 -> no bank-aligned rows

typedef __bf16 bf16_t;
typedef __attribute__((ext_vector_type(16))) bf16_t v16bf;
typedef __attribute__((ext_vector_type(8)))  bf16_t v8bf;
typedef __attribute__((ext_vector_type(8)))  float  v8f;
typedef int v4i_gcc __attribute__((vector_size(16)));   // matches async-LDS builtin param type

struct U4 { unsigned int x, y, z, w; };
struct U8 { U4 lo, hi; };
struct F4 { float x, y, z, w; };
struct S4 { unsigned short x, y, z, w; };

#if defined(__has_builtin)
#if __has_builtin(__builtin_amdgcn_global_load_async_to_lds_b128) && \
    __has_builtin(__builtin_amdgcn_s_wait_asynccnt)
#define GRU_ASYNC_LDS 1
#endif
#endif

// 16-byte global -> LDS copy: async (ASYNCcnt-tracked) on CDNA5 toolchains, sync fallback.
__device__ __forceinline__ void copy16_g2l(const unsigned short* g, unsigned short* l) {
#if defined(GRU_ASYNC_LDS)
    __builtin_amdgcn_global_load_async_to_lds_b128((v4i_gcc*)g, (v4i_gcc*)l, 0, 0);
#else
    *(U4*)l = *(const U4*)g;
#endif
}
__device__ __forceinline__ void wait_lds_copies() {
#if defined(GRU_ASYNC_LDS)
    __builtin_amdgcn_s_wait_asynccnt(0);
#endif
}

__device__ __forceinline__ unsigned short f2bf(float f) {
    unsigned int u = __float_as_uint(f);
    u += 0x7FFFu + ((u >> 16) & 1u);   // round-to-nearest-even
    return (unsigned short)(u >> 16);
}

// A-fragment (16x32 bf16): lane l -> row given, element e: K = (l/16)*16 + e
__device__ __forceinline__ v16bf load_a_frag(const unsigned short* base, int row_stride,
                                             int row, int kc, int lane) {
    const int half = lane >> 4;
    const unsigned short* p0 = base + row * row_stride + kc + half * 16;
    v8bf lo = __builtin_bit_cast(v8bf, *(const U4*)p0);
    v8bf hi = __builtin_bit_cast(v8bf, *(const U4*)(p0 + 8));
    v16bf a;
#pragma unroll
    for (int i = 0; i < 8; ++i) { a[i] = lo[i]; a[i + 8] = hi[i]; }
    return a;
}

// B-fragment (32x16 bf16), B[k][n] = W[n][k], W row-major with row length 512:
// lane l -> col N = n0 + (l&15), K block = kc + (l>>4)*16, 16 consecutive k (32B load).
__device__ __forceinline__ v16bf load_b_frag(const unsigned short* w, int n0, int kc, int lane) {
    const unsigned short* p = w + (n0 + (lane & 15)) * 512 + kc + (lane >> 4) * 16;
    return __builtin_bit_cast(v16bf, *(const U8*)p);
}

#define WMMA_BF16(a, b, c) \
    __builtin_amdgcn_wmma_f32_16x16x32_bf16(false, (a), false, (b), (short)0, (c), false, false)

__global__ __launch_bounds__(256)
void cvt_f32_bf16(const float* __restrict__ src, unsigned short* __restrict__ dst, int n) {
    int i4 = (blockIdx.x * blockDim.x + threadIdx.x) * 4;
    if (i4 < n) {
        F4 f = *(const F4*)(src + i4);
        S4 s = { f2bf(f.x), f2bf(f.y), f2bf(f.z), f2bf(f.w) };
        *(S4*)(dst + i4) = s;
    }
}

// Input-side GEMM, all 3 gates per block against one LDS-staged A tile.
// grid = (65536/32, 512/256), block = 256 (8 waves); wave: 16 rows x 64 cols per gate.
__global__ __launch_bounds__(256)
void gru_input_gemm(const unsigned short* __restrict__ xb,
                    const unsigned short* __restrict__ wb,   // 3 input weights, bf16
                    float* __restrict__ G) {
    __shared__ unsigned short xA[32 * LDA];

    const int tid  = threadIdx.x;
    const int lane = tid & 31;
    const int w    = tid >> 5;
    const int m0b  = blockIdx.x * 32;
    const int nb   = blockIdx.y * 256 + (w & 3) * 64;
    const int mw   = (w >> 2) * 16;

    // Stage A tile (32 rows x 512 K, bf16) into LDS via async copies.
    {
        const unsigned short* gsrc = xb + (size_t)m0b * I_;
#pragma unroll
        for (int j = 0; j < 8; ++j) {
            int c   = tid + 256 * j;          // 2048 x 16-byte chunks
            int row = c >> 6;
            int cc  = (c & 63) * 8;
            copy16_g2l(gsrc + row * I_ + cc, &xA[row * LDA + cc]);
        }
        wait_lds_copies();
    }
    __syncthreads();

    const int arow = mw + (lane & 15);
    const int half = lane >> 4;
    const int col  = lane & 15;
    const size_t BTH = (size_t)(B_ * T_) * H_;

    for (int g = 0; g < 3; ++g) {
        const unsigned short* wg = wb + (size_t)g * (H_ * I_);
        float* Gg = G + (size_t)g * BTH;
        v8f acc0 = {}, acc1 = {}, acc2 = {}, acc3 = {};
        // software pipeline: B fragments loaded one K-chunk ahead of their WMMA
        v16bf b0 = load_b_frag(wg, nb +  0, 0, lane);
        v16bf b1 = load_b_frag(wg, nb + 16, 0, lane);
        v16bf b2 = load_b_frag(wg, nb + 32, 0, lane);
        v16bf b3 = load_b_frag(wg, nb + 48, 0, lane);
        for (int kc = 0; kc < I_; kc += 32) {
            v16bf a = load_a_frag(xA, LDA, arow, kc, lane);
            const int kn = (kc + 32) & (I_ - 1);
            v16bf p0 = load_b_frag(wg, nb +  0, kn, lane);
            v16bf p1 = load_b_frag(wg, nb + 16, kn, lane);
            v16bf p2 = load_b_frag(wg, nb + 32, kn, lane);
            v16bf p3 = load_b_frag(wg, nb + 48, kn, lane);
            acc0 = WMMA_BF16(a, b0, acc0);
            acc1 = WMMA_BF16(a, b1, acc1);
            acc2 = WMMA_BF16(a, b2, acc2);
            acc3 = WMMA_BF16(a, b3, acc3);
            b0 = p0; b1 = p1; b2 = p2; b3 = p3;
        }
#pragma unroll
        for (int v = 0; v < 8; ++v) {
            size_t row = (size_t)(m0b + mw + v + 8 * half);
            Gg[row * H_ + nb +  0 + col] = acc0[v];
            Gg[row * H_ + nb + 16 + col] = acc1[v];
            Gg[row * H_ + nb + 32 + col] = acc2[v];
            Gg[row * H_ + nb + 48 + col] = acc3[v];
        }
    }
}

// Persistent recurrence: recurrence is independent per batch row -> one block per 16 batch
// rows iterates all T steps with only intra-workgroup barriers. 32 waves; wave w owns
// columns [16w, 16w+16). h lives in LDS (f32 master + bf16 matmul copy).
__global__ __launch_bounds__(1024)
void gru_scan(const float* __restrict__ G,
              const unsigned short* __restrict__ whb,  // bf16 W_hr, W_hz, W_hn
              const float* __restrict__ h0,
              const float* __restrict__ b_hr,
              const float* __restrict__ b_hz,
              const float* __restrict__ b_hn,
              float* __restrict__ out,
              float* __restrict__ h_n) {
    __shared__ unsigned short hA[16 * LDA];
    __shared__ unsigned short rhA[16 * LDA];
    __shared__ float          hF[16 * H_];

    const int tid  = threadIdx.x;
    const int lane = tid & 31;
    const int w    = tid >> 5;
    const int n0   = w * 16;
    const int b0   = blockIdx.x * 16;
    const int col  = lane & 15;
    const int half = lane >> 4;

    const unsigned short* Whr = whb;
    const unsigned short* Whz = whb + H_ * H_;
    const unsigned short* Whn = whb + 2 * H_ * H_;

    const size_t BTH = (size_t)(B_ * T_) * H_;
    const float* Gr = G;
    const float* Gz = G + BTH;
    const float* Gn = G + 2 * BTH;

    const float bhr = b_hr[n0 + col];
    const float bhz = b_hz[n0 + col];
    const float bhn = b_hn[n0 + col];

    for (int i = tid; i < 16 * H_; i += 1024) {
        int row = i >> 9, c = i & (H_ - 1);
        float v = h0[(b0 + row) * H_ + c];
        hF[row * H_ + c] = v;
        hA[row * LDA + c] = f2bf(v);
    }
    __syncthreads();

    for (int t = 0; t < T_; ++t) {
        // hoist HBM gate loads so the WMMA K-loop hides their latency
        float gr[8], gz[8];
#pragma unroll
        for (int v = 0; v < 8; ++v) {
            int row   = v + 8 * half;
            size_t gi = ((size_t)(b0 + row) * T_ + t) * H_ + n0 + col;
            gr[v] = Gr[gi];
            gz[v] = Gz[gi];
        }

        // ---- r, z gates: h @ W_hr^T, h @ W_hz^T (B frags pipelined one chunk ahead) ----
        v8f ar = {}, az = {};
        v16bf br = load_b_frag(Whr, n0, 0, lane);
        v16bf bz = load_b_frag(Whz, n0, 0, lane);
        for (int kc = 0; kc < H_; kc += 32) {
            v16bf a = load_a_frag(hA, LDA, (lane & 15), kc, lane);
            const int kn = (kc + 32) & (H_ - 1);
            v16bf pr = load_b_frag(Whr, n0, kn, lane);
            v16bf pz = load_b_frag(Whz, n0, kn, lane);
            ar = WMMA_BF16(a, br, ar);
            az = WMMA_BF16(a, bz, az);
            br = pr; bz = pz;
        }
        float zv[8];
#pragma unroll
        for (int v = 0; v < 8; ++v) {
            int row = v + 8 * half;
            float r = 1.f / (1.f + __expf(-(gr[v] + ar[v] + bhr)));
            float z = 1.f / (1.f + __expf(-(gz[v] + az[v] + bhz)));
            zv[v] = z;
            float rh = r * hF[row * H_ + n0 + col];
            rhA[row * LDA + n0 + col] = f2bf(rh);
        }
        __syncthreads();

        // ---- n gate: (r*h) @ W_hn^T ----
        float gn[8];
#pragma unroll
        for (int v = 0; v < 8; ++v) {
            int row   = v + 8 * half;
            size_t gi = ((size_t)(b0 + row) * T_ + t) * H_ + n0 + col;
            gn[v] = Gn[gi];
        }
        v8f an = {};
        v16bf bn = load_b_frag(Whn, n0, 0, lane);
        for (int kc = 0; kc < H_; kc += 32) {
            v16bf a = load_a_frag(rhA, LDA, (lane & 15), kc, lane);
            const int kn = (kc + 32) & (H_ - 1);
            v16bf pn = load_b_frag(Whn, n0, kn, lane);
            an = WMMA_BF16(a, bn, an);
            bn = pn;
        }
#pragma unroll
        for (int v = 0; v < 8; ++v) {
            int row   = v + 8 * half;
            size_t gi = ((size_t)(b0 + row) * T_ + t) * H_ + n0 + col;
            float nn   = tanhf(gn[v] + an[v] + bhn);
            float hold = hF[row * H_ + n0 + col];
            float hnew = (1.f - zv[v]) * hold + zv[v] * nn;
            out[gi] = hnew;                       // out is (B,T,H): same flat index
            hF[row * H_ + n0 + col] = hnew;
            hA[row * LDA + n0 + col] = f2bf(hnew);
        }
        // prefetch next step's gate rows into cache (global_prefetch_b8)
        if (t + 1 < T_) {
#pragma unroll
            for (int v = 0; v < 8; ++v) {
                int row   = v + 8 * half;
                size_t gi = ((size_t)(b0 + row) * T_ + (t + 1)) * H_ + n0 + col;
                __builtin_prefetch(&Gr[gi], 0, 3);
                __builtin_prefetch(&Gz[gi], 0, 3);
                __builtin_prefetch(&Gn[gi], 0, 3);
            }
        }
        __syncthreads();
    }

#pragma unroll
    for (int v = 0; v < 8; ++v) {
        int row = v + 8 * half;
        h_n[(size_t)(b0 + row) * H_ + n0 + col] = hF[row * H_ + n0 + col];
    }
}

extern "C" void kernel_launch(void* const* d_in, const int* in_sizes, int n_in,
                              void* d_out, int out_size, void* d_ws, size_t ws_size,
                              hipStream_t stream) {
    (void)in_sizes; (void)n_in; (void)out_size; (void)ws_size;
    const float* x   = (const float*)d_in[0];
    const float* h0  = (const float*)d_in[1];
    const float* Wi[3] = { (const float*)d_in[2], (const float*)d_in[3], (const float*)d_in[4] };
    const float* Wh[3] = { (const float*)d_in[5], (const float*)d_in[6], (const float*)d_in[7] };
    const float* bhr = (const float*)d_in[8];
    const float* bhz = (const float*)d_in[9];
    const float* bhn = (const float*)d_in[10];

    // workspace: bf16 x | bf16 weights (3 input, 3 hidden) | f32 gate pre-activations
    unsigned short* xb = (unsigned short*)d_ws;                       // B*T*I bf16
    unsigned short* wb = xb + (size_t)B_ * T_ * I_;                   // 6*H*I bf16
    float*          G  = (float*)(wb + 6 * (size_t)(H_ * I_));        // 3*B*T*H f32

    int n = B_ * T_ * I_;
    cvt_f32_bf16<<<(n / 4 + 255) / 256, 256, 0, stream>>>(x, xb, n);
    for (int g = 0; g < 3; ++g) {
        cvt_f32_bf16<<<(H_ * I_ / 4 + 255) / 256, 256, 0, stream>>>(Wi[g], wb + (size_t)g * H_ * I_, H_ * I_);
        cvt_f32_bf16<<<(H_ * I_ / 4 + 255) / 256, 256, 0, stream>>>(Wh[g], wb + (size_t)(3 + g) * H_ * I_, H_ * I_);
    }

    gru_input_gemm<<<dim3((B_ * T_) / 32, H_ / 256), 256, 0, stream>>>(xb, wb, G);

    float* out = (float*)d_out;
    float* h_n = out + (size_t)B_ * T_ * H_;
    gru_scan<<<dim3(B_ / 16), 1024, 0, stream>>>(G, wb + 3 * (size_t)(H_ * I_), h0,
                                                 bhr, bhz, bhn, out, h_n);
}
